// Edgeconvfc_687194767627
// MI455X (gfx1250) — compile-verified, matrix operand order checked by reading, not run on hardware
//
#include <hip/hip_runtime.h>
#include <hip/hip_bf16.h>

#define N_NODES 100000
#define N_EDGES 1000000
#define N_IN 128
#define N_HID 64
#define E_IN 16
#define EF 16
#define E_HID 64
#define IN_CH (N_HID + E_IN + EF)   // 96

typedef __attribute__((ext_vector_type(16))) __bf16 v16bf;
typedef __attribute__((ext_vector_type(8)))  float  v8f;
typedef __attribute__((ext_vector_type(4)))  float  v4f;

// K index inside a 16x32 bf16 A/B fragment for vector element e (0..15) of a lane.
// lanes 0-15 hold K = {0..7, 16..23}; lanes 16-31 hold K = {8..15, 24..31}.
__device__ __forceinline__ int frag_k(int e, int laneHi) {
    return (e & 7) | ((e >> 3) << 4) | (laneHi << 3);
}

// ---------------------------------------------------------------------------
// start = min(edge_index[0])
// ---------------------------------------------------------------------------
__global__ void init_min_kernel(int* p) {
    if (threadIdx.x == 0) *p = 0x7fffffff;
}

__global__ __launch_bounds__(256) void min_reduce_kernel(const int* __restrict__ ei,
                                                         int n, int* __restrict__ p) {
    int v = 0x7fffffff;
    for (int i = blockIdx.x * blockDim.x + threadIdx.x; i < n;
         i += gridDim.x * blockDim.x) {
        int t = ei[i];
        v = t < v ? t : v;
    }
    atomicMin(p, v);
}

// ---------------------------------------------------------------------------
// h = x @ W1^T + b1   ([N_NODES,128] x [128,64] -> f32)
// W1 staged in LDS as bf16 in WMMA B-fragment layout:
//   frag = kt*4 + nt (kt: K-tile of 32, nt: N-tile of 16)
//   ldsW[frag*512 + lane*16 + e] = W1[n][k],  n = nt*16 + lane%16,
//                                  k = kt*32 + frag_k(e, lane/16)
// ---------------------------------------------------------------------------
__global__ __launch_bounds__(256) void node_linear_kernel(
    const float* __restrict__ x, const float* __restrict__ W1,
    const float* __restrict__ b1, float* __restrict__ h) {
    __shared__ __bf16 ldsW[16 * 512];          // 16 KB

    const int lane   = threadIdx.x & 31;
    const int wave   = threadIdx.x >> 5;
    const int laneHi = lane >> 4;

    for (int p = threadIdx.x; p < 16 * 32; p += blockDim.x) {
        const int frag = p >> 5;
        const int l    = p & 31;
        const int kt   = frag >> 2;
        const int nt   = frag & 3;
        const int lHi  = l >> 4;
        const int n    = (l & 15) + nt * 16;
#pragma unroll
        for (int e = 0; e < 16; ++e) {
            const int k = 32 * kt + frag_k(e, lHi);
            ldsW[frag * 512 + l * 16 + e] = (__bf16)W1[n * N_IN + k];
        }
    }
    __syncthreads();

    float biasCol[4];
#pragma unroll
    for (int nt = 0; nt < 4; ++nt) biasCol[nt] = b1[nt * 16 + (lane & 15)];

    const int nTiles = (N_NODES + 15) / 16;    // 6250
    for (int tile = blockIdx.x * 8 + wave; tile < nTiles; tile += gridDim.x * 8) {
        const int m0 = tile * 16;
        int m = m0 + (lane & 15);
        if (m >= N_NODES) m = N_NODES - 1;     // clamp; stores are guarded
        const float* __restrict__ xrow = x + (size_t)m * N_IN;

        v8f acc[4];
#pragma unroll
        for (int nt = 0; nt < 4; ++nt)
#pragma unroll
            for (int j = 0; j < 8; ++j) acc[nt][j] = 0.0f;

#pragma unroll
        for (int kt = 0; kt < 4; ++kt) {
            const int base = 32 * kt + laneHi * 8;
            const v4f r0 = *(const v4f*)(xrow + base);
            const v4f r1 = *(const v4f*)(xrow + base + 4);
            const v4f r2 = *(const v4f*)(xrow + base + 16);
            const v4f r3 = *(const v4f*)(xrow + base + 20);
            v16bf a;
#pragma unroll
            for (int j = 0; j < 4; ++j) {
                a[j]      = (__bf16)r0[j];
                a[4 + j]  = (__bf16)r1[j];
                a[8 + j]  = (__bf16)r2[j];
                a[12 + j] = (__bf16)r3[j];
            }
#pragma unroll
            for (int nt = 0; nt < 4; ++nt) {
                const v16bf b =
                    *(const v16bf*)(&ldsW[(kt * 4 + nt) * 512 + lane * 16]);
                acc[nt] = __builtin_amdgcn_wmma_f32_16x16x32_bf16(
                    false, a, false, b, (short)0, acc[nt], false, false);
            }
        }

#pragma unroll
        for (int nt = 0; nt < 4; ++nt) {
#pragma unroll
            for (int r = 0; r < 8; ++r) {
                const int row = m0 + r + laneHi * 8;
                if (row < N_NODES)
                    h[(size_t)row * N_HID + nt * 16 + (lane & 15)] =
                        acc[nt][r] + biasCol[nt];
            }
        }
    }
}

// ---------------------------------------------------------------------------
// out = [relu(h[src]-h[dst]) | edge_attr | edge_f] @ W2^T + b2
// ---------------------------------------------------------------------------
__global__ __launch_bounds__(256) void edge_mlp_kernel(
    const float* __restrict__ h, const int* __restrict__ edge_index,
    const float* __restrict__ edge_attr, const float* __restrict__ edge_f,
    const float* __restrict__ W2, const float* __restrict__ b2,
    const int* __restrict__ minp, float* __restrict__ out) {
    __shared__ __bf16 ldsW[12 * 512];          // 12 KB

    const int lane   = threadIdx.x & 31;
    const int wave   = threadIdx.x >> 5;
    const int laneHi = lane >> 4;
    const int start  = *minp;

    for (int p = threadIdx.x; p < 12 * 32; p += blockDim.x) {
        const int frag = p >> 5;
        const int l    = p & 31;
        const int kt   = frag >> 2;
        const int nt   = frag & 3;
        const int lHi  = l >> 4;
        const int n    = (l & 15) + nt * 16;
#pragma unroll
        for (int e = 0; e < 16; ++e) {
            const int k = 32 * kt + frag_k(e, lHi);
            ldsW[frag * 512 + l * 16 + e] = (__bf16)W2[n * IN_CH + k];
        }
    }
    __syncthreads();

    float biasCol[4];
#pragma unroll
    for (int nt = 0; nt < 4; ++nt) biasCol[nt] = b2[nt * 16 + (lane & 15)];

    const int nTiles = (N_EDGES + 15) / 16;    // 62500
    const int tStride = gridDim.x * 8;
    for (int tile = blockIdx.x * 8 + wave; tile < nTiles; tile += tStride) {
        const int m0 = tile * 16;
        int em = m0 + (lane & 15);
        if (em >= N_EDGES) em = N_EDGES - 1;   // clamp; stores are guarded

        // Prefetch next tile's edge indices (global_prefetch_b8).
        const int nextEm = em + tStride * 16;
        if (nextEm < N_EDGES) {
            __builtin_prefetch(edge_index + nextEm, 0, 1);
            __builtin_prefetch(edge_index + N_EDGES + nextEm, 0, 1);
        }

        const int srow = edge_index[em] - start;
        const int drow = edge_index[N_EDGES + em] - start;
        const float* __restrict__ hs = h + (size_t)srow * N_HID;
        const float* __restrict__ hd = h + (size_t)drow * N_HID;

        v8f acc[4];
#pragma unroll
        for (int nt = 0; nt < 4; ++nt)
#pragma unroll
            for (int j = 0; j < 8; ++j) acc[nt][j] = 0.0f;

        // K-tiles 0,1: fused gather + sub + relu + bf16 quantize
#pragma unroll
        for (int kt = 0; kt < 2; ++kt) {
            const int base = 32 * kt + laneHi * 8;
            v16bf a;
#pragma unroll
            for (int half = 0; half < 2; ++half) {
                const int kb = base + half * 16;
                const v4f s0 = *(const v4f*)(hs + kb);
                const v4f s1 = *(const v4f*)(hs + kb + 4);
                const v4f d0 = *(const v4f*)(hd + kb);
                const v4f d1 = *(const v4f*)(hd + kb + 4);
#pragma unroll
                for (int j = 0; j < 4; ++j) {
                    float v0 = s0[j] - d0[j]; v0 = v0 > 0.0f ? v0 : 0.0f;
                    float v1 = s1[j] - d1[j]; v1 = v1 > 0.0f ? v1 : 0.0f;
                    a[half * 8 + j]     = (__bf16)v0;
                    a[half * 8 + 4 + j] = (__bf16)v1;
                }
            }
#pragma unroll
            for (int nt = 0; nt < 4; ++nt) {
                const v16bf b =
                    *(const v16bf*)(&ldsW[(kt * 4 + nt) * 512 + lane * 16]);
                acc[nt] = __builtin_amdgcn_wmma_f32_16x16x32_bf16(
                    false, a, false, b, (short)0, acc[nt], false, false);
            }
        }

        // K-tile 2: feat cols 64..95 = edge_attr | edge_f
        {
            const int base2 = laneHi * 8;
            const float* ea = edge_attr + (size_t)em * E_IN + base2;
            const float* ef = edge_f    + (size_t)em * EF   + base2;
            const v4f e0 = *(const v4f*)(ea);
            const v4f e1 = *(const v4f*)(ea + 4);
            const v4f f0 = *(const v4f*)(ef);
            const v4f f1 = *(const v4f*)(ef + 4);
            v16bf a;
#pragma unroll
            for (int j = 0; j < 4; ++j) {
                a[j]      = (__bf16)e0[j];
                a[4 + j]  = (__bf16)e1[j];
                a[8 + j]  = (__bf16)f0[j];
                a[12 + j] = (__bf16)f1[j];
            }
#pragma unroll
            for (int nt = 0; nt < 4; ++nt) {
                const v16bf b =
                    *(const v16bf*)(&ldsW[(2 * 4 + nt) * 512 + lane * 16]);
                acc[nt] = __builtin_amdgcn_wmma_f32_16x16x32_bf16(
                    false, a, false, b, (short)0, acc[nt], false, false);
            }
        }

#pragma unroll
        for (int nt = 0; nt < 4; ++nt) {
#pragma unroll
            for (int r = 0; r < 8; ++r) {
                const int row = m0 + r + laneHi * 8;
                if (row < N_EDGES)
                    out[(size_t)row * E_HID + nt * 16 + (lane & 15)] =
                        acc[nt][r] + biasCol[nt];
            }
        }
    }
}

// ---------------------------------------------------------------------------
extern "C" void kernel_launch(void* const* d_in, const int* in_sizes, int n_in,
                              void* d_out, int out_size, void* d_ws, size_t ws_size,
                              hipStream_t stream) {
    // setup_inputs() order: x, edge_index, edge_f, edge_attr, W1, b1, W2, b2
    const float* x          = (const float*)d_in[0];
    const int*   edge_index = (const int*)  d_in[1];
    const float* edge_f     = (const float*)d_in[2];
    const float* edge_attr  = (const float*)d_in[3];
    const float* W1         = (const float*)d_in[4];
    const float* b1         = (const float*)d_in[5];
    const float* W2         = (const float*)d_in[6];
    const float* b2         = (const float*)d_in[7];
    float*       out        = (float*)d_out;

    // workspace: h (f32, 25.6 MB) then the min scalar
    float* h    = (float*)d_ws;
    int*   minp = (int*)((char*)d_ws + (size_t)N_NODES * N_HID * sizeof(float));

    init_min_kernel<<<1, 32, 0, stream>>>(minp);
    min_reduce_kernel<<<256, 256, 0, stream>>>(edge_index, N_EDGES, minp);

    // node GEMM: 6250 M-tiles, 8 waves/block -> 782 blocks covers in one pass
    node_linear_kernel<<<782, 256, 0, stream>>>(x, W1, b1, h);

    // edge GEMM: 62500 tiles; persistent-ish grid, ~4 tiles per wave
    edge_mlp_kernel<<<2048, 256, 0, stream>>>(h, edge_index, edge_attr, edge_f,
                                              W2, b2, minp, out);
}
